// SpMiddleNoDownsampleXYNoExpand_4750233829985
// MI455X (gfx1250) — matrix operand order repaired and checked by reading, not compiled, not run on hardware
//
#include <hip/hip_runtime.h>
#include <hip/hip_bf16.h>

// ---------------------------------------------------------------------------
// Types for CDNA5 WMMA (wave32, 16x16x32 bf16 -> f32)
// ---------------------------------------------------------------------------
typedef __attribute__((ext_vector_type(16))) __bf16 v16bf;
typedef __attribute__((ext_vector_type(8)))  float  v8f;

union Frag { unsigned u[8]; v16bf v; };

__device__ __forceinline__ unsigned short f2bf(float f) {
    union { float f; unsigned u; } a; a.f = f;
    unsigned u = a.u;
    unsigned r = u + 0x7FFFu + ((u >> 16) & 1u);   // round-to-nearest-even
    return (unsigned short)(r >> 16);
}
__device__ __forceinline__ float bf2f(unsigned short h) {
    union { unsigned u; float f; } a; a.u = ((unsigned)h) << 16;
    return a.f;
}

// ---------------------------------------------------------------------------
// Utility kernels
// ---------------------------------------------------------------------------
__global__ void zero_u32(unsigned* __restrict__ p, long long n) {
    long long i = (long long)blockIdx.x * blockDim.x + threadIdx.x;
    long long stride = (long long)gridDim.x * blockDim.x;
    for (; i < n; i += stride) p[i] = 0u;
}

__global__ void scatter_kernel(const float* __restrict__ vf, const int* __restrict__ coors,
                               unsigned short* __restrict__ X0, unsigned char* __restrict__ Msk,
                               int N) {
    int n = blockIdx.x * blockDim.x + threadIdx.x;
    if (n >= N) return;
    int z = coors[n * 4 + 1];
    int y = coors[n * 4 + 2];
    int x = coors[n * 4 + 3];
    if (z < 0 || z >= 41 || y < 0 || y >= 160 || x < 0 || x >= 160) return;
    size_t p = ((size_t)(z * 160 + y) * 160 + x);
    Msk[p] = 1;
#pragma unroll
    for (int c = 0; c < 16; ++c)
        X0[p * 32 + c] = f2bf(vf[(size_t)n * 16 + c]);
    // channels 16..31 already zeroed
}

// Convert DHWIO f32 weights -> bf16 [cout][tap*cin_pad + cin] (zero-padded cin)
__global__ void prep_w(const float* __restrict__ w, unsigned short* __restrict__ wt,
                       int cin_real, int cin_pad, int cout) {
    int idx = blockIdx.x * blockDim.x + threadIdx.x;
    int total = cout * 27 * cin_pad;
    if (idx >= total) return;
    int co  = idx / (27 * cin_pad);
    int rem = idx - co * (27 * cin_pad);
    int tap = rem / cin_pad;
    int ci  = rem - tap * cin_pad;
    float val = (ci < cin_real) ? w[((size_t)tap * cin_real + ci) * cout + co] : 0.f;
    wt[idx] = f2bf(val);
}

// ---------------------------------------------------------------------------
// Implicit-GEMM 3x3x3 conv with WMMA bf16 + fused BN/ReLU/mask epilogue.
// One block = ROWS output rows (W=160) at fixed (d, h0). 128 threads = 4
// waves; each wave owns an N-tile PAIR (2x16 couts) and 5 M-tiles, so every
// A fragment (LDS) is reused by two WMMAs -> half the LDS traffic per WMMA.
// ---------------------------------------------------------------------------
template<int CIN, int COUT, int ROWS>
__global__ __launch_bounds__(128) void conv3d_wmma(
    const unsigned short* __restrict__ X,    // bf16 [D][H][W][CIN]
    const unsigned short* __restrict__ Wt,   // bf16 [COUT][27*CIN]
    const unsigned char*  __restrict__ Msk,  // [D][H][W]
    unsigned short*       __restrict__ Y,    // bf16 [D][H][W][COUT]
    int Dsz,
    const float* __restrict__ bg, const float* __restrict__ bb,
    const float* __restrict__ bm, const float* __restrict__ bv) {
    constexpr int W = 160, H = 160, W2 = 162;
    constexpr int CIND = (CIN + 2) / 2;           // dwords per LDS position (padded)
    constexpr int NT  = COUT / 16;
    constexpr int NTP = NT / 2;                   // N-tile pairs
    constexpr int MT  = ROWS * 10;
    static_assert(MT * NT == 40, "4 waves x (2 ntiles x 5 mtiles)");

    __shared__ unsigned smem[(ROWS + 2) * W2 * CIND];

    const int tid   = threadIdx.x;                // 0..127
    const int wave  = tid >> 5;                   // 0..3
    const int lane  = tid & 31;
    const int half  = lane >> 4;
    const int ln16  = lane & 15;
    const int pair  = wave % NTP;                 // which ntile pair
    const int mbase = (wave / NTP) * 5;           // first of 5 mtiles
    const int hblocks = H / ROWS;
    const int d  = blockIdx.x / hblocks;
    const int h0 = (blockIdx.x % hblocks) * ROWS;

    const v8f vzero = {0.f, 0.f, 0.f, 0.f, 0.f, 0.f, 0.f, 0.f};
    v8f acc[2][5];
#pragma unroll
    for (int j = 0; j < 2; ++j)
#pragma unroll
        for (int i = 0; i < 5; ++i) acc[j][i] = vzero;

    const unsigned* Wd = (const unsigned*)Wt;
    const int cout0 = pair * 32 + ln16;           // ntile 2*pair
    const int cout1 = cout0 + 16;                 // ntile 2*pair+1

    for (int dz = 0; dz < 3; ++dz) {
        __syncthreads();
        // ---- cooperative stage of (ROWS+2) halo rows at depth d+dz-1 ----
        const int dd = d + dz - 1;
        const int totalDw = (ROWS + 2) * W2 * (CIN / 2);
        for (int t = tid; t < totalDw; t += 128) {
            int pos = t / (CIN / 2);
            int cp  = t - pos * (CIN / 2);
            int rr  = pos / W2;
            int wi  = pos - rr * W2;
            int hh = h0 - 1 + rr;
            int ww = wi - 1;
            unsigned val = 0;
            if (dd >= 0 && dd < Dsz && hh >= 0 && hh < H && ww >= 0 && ww < W)
                val = *(const unsigned*)(X + ((size_t)(dd * H + hh) * W + ww) * CIN + cp * 2);
            smem[pos * CIND + cp] = val;
        }
        __syncthreads();

#pragma unroll
        for (int dy = 0; dy < 3; ++dy) {
#pragma unroll
            for (int dx = 0; dx < 3; ++dx) {
#pragma unroll
                for (int kb = 0; kb < CIN / 32; ++kb) {
                    const int tap = (dz * 3 + dy) * 3 + dx;
                    // ---- two B fragments (one per ntile of the pair) ----
                    Frag bf0, bf1;
                    {
                        const int koff = tap * (CIN / 2) + kb * 16 + half * 8;
                        const int b0 = cout0 * (27 * CIN / 2) + koff;
                        const int b1 = cout1 * (27 * CIN / 2) + koff;
#pragma unroll
                        for (int vv = 0; vv < 8; ++vv) bf0.u[vv] = Wd[b0 + vv];
#pragma unroll
                        for (int vv = 0; vv < 8; ++vv) bf1.u[vv] = Wd[b1 + vv];
                    }
#pragma unroll
                    for (int i = 0; i < 5; ++i) {
                        const int mtile = mbase + i;
                        const int r    = mtile / 10;
                        const int wseg = (mtile - r * 10) * 16;
                        // ---- A fragment (LDS), shared by both WMMAs ----
                        Frag af;
                        const int abase = ((dy + r) * W2 + (wseg + ln16 + dx)) * CIND
                                        + kb * 16 + 8 * half;
#pragma unroll
                        for (int vv = 0; vv < 4; ++vv) af.u[vv] = smem[abase + vv];
#pragma unroll
                        for (int vv = 4; vv < 8; ++vv) af.u[vv] = smem[abase + vv];
                        acc[0][i] = __builtin_amdgcn_wmma_f32_16x16x32_bf16(
                            false, af.v, false, bf0.v, (short)0, acc[0][i], false, false);
                        acc[1][i] = __builtin_amdgcn_wmma_f32_16x16x32_bf16(
                            false, af.v, false, bf1.v, (short)0, acc[1][i], false, false);
                    }
                }
            }
        }
    }

    // ---- fused BN + ReLU + mask epilogue, store bf16 ----
#pragma unroll
    for (int j = 0; j < 2; ++j) {
        const int coutIdx = (j == 0) ? cout0 : cout1;
        const float scl = bg[coutIdx] * rsqrtf(bv[coutIdx] + 1e-3f);
        const float bia = bb[coutIdx] - bm[coutIdx] * scl;
#pragma unroll
        for (int i = 0; i < 5; ++i) {
            const int mtile = mbase + i;
            const int r    = mtile / 10;
            const int wseg = (mtile - r * 10) * 16;
            const int hh = h0 + r;
#pragma unroll
            for (int rv = 0; rv < 8; ++rv) {
                const int M  = rv + 8 * half;
                const int ww = wseg + M;
                const size_t p = (size_t)(d * H + hh) * W + ww;
                float y = acc[j][i][rv] * scl + bia;
                y = (Msk[p] != 0) ? fmaxf(y, 0.f) : 0.f;
                Y[p * COUT + coutIdx] = f2bf(y);
            }
        }
    }
}

// ---------------------------------------------------------------------------
// Masked max-pool (3,3,3) stride (2,1,1), depth pad PD, + fused BN/ReLU.
// ---------------------------------------------------------------------------
template<int PD>
__global__ void pool_bn(const unsigned short* __restrict__ X, const unsigned char* __restrict__ Min,
                        unsigned short* __restrict__ Y, unsigned char* __restrict__ Mout,
                        int ID, int OD,
                        const float* __restrict__ g, const float* __restrict__ b,
                        const float* __restrict__ m, const float* __restrict__ v) {
    constexpr int W = 160, H = 160, C = 64;
    long long idx = (long long)blockIdx.x * blockDim.x + threadIdx.x;
    long long total = (long long)OD * H * W * C;
    if (idx >= total) return;
    int c = (int)(idx & 63);
    long long rem = idx >> 6;
    int w = (int)(rem % W); rem /= W;
    int h = (int)(rem % H);
    int od = (int)(rem / H);

    float mx = -1e30f;
    bool any = false;
#pragma unroll
    for (int kz = 0; kz < 3; ++kz) {
        int zd = od * 2 - PD + kz;
        if (zd < 0 || zd >= ID) continue;
#pragma unroll
        for (int ky = 0; ky < 3; ++ky) {
            int hy = h + ky - 1;
            if (hy < 0 || hy >= H) continue;
#pragma unroll
            for (int kx = 0; kx < 3; ++kx) {
                int wx = w + kx - 1;
                if (wx < 0 || wx >= W) continue;
                size_t p = (size_t)(zd * H + hy) * W + wx;
                if (Min[p]) {
                    any = true;
                    mx = fmaxf(mx, bf2f(X[p * C + c]));
                }
            }
        }
    }
    float scl = g[c] * rsqrtf(v[c] + 1e-3f);
    float bia = b[c] - m[c] * scl;
    float v0 = any ? mx : 0.f;
    float y  = any ? fmaxf(v0 * scl + bia, 0.f) : 0.f;
    size_t po = (size_t)(od * H + h) * W + w;
    Y[po * C + c] = f2bf(y);
    if (c == 0) Mout[po] = any ? 1 : 0;
}

// Final pool (3,1,1) stride (2,1,1) pad 0 over D=10 -> 4, fused BN/ReLU,
// write f32 output in (c*4+od, h, w) layout.
__global__ void pool3_out(const unsigned short* __restrict__ X, const unsigned char* __restrict__ Msk,
                          float* __restrict__ Out,
                          const float* __restrict__ g, const float* __restrict__ b,
                          const float* __restrict__ m, const float* __restrict__ v) {
    constexpr int W = 160, H = 160, C = 64, ID = 10, OD = 4;
    long long idx = (long long)blockIdx.x * blockDim.x + threadIdx.x;
    long long total = (long long)OD * H * W * C;
    if (idx >= total) return;
    int c = (int)(idx & 63);
    long long rem = idx >> 6;
    int w = (int)(rem % W); rem /= W;
    int h = (int)(rem % H);
    int od = (int)(rem / H);

    float mx = -1e30f;
    bool any = false;
#pragma unroll
    for (int kz = 0; kz < 3; ++kz) {
        int zd = od * 2 + kz;
        if (zd >= ID) continue;
        size_t p = (size_t)(zd * H + h) * W + w;
        if (Msk[p]) {
            any = true;
            mx = fmaxf(mx, bf2f(X[p * C + c]));
        }
    }
    float scl = g[c] * rsqrtf(v[c] + 1e-3f);
    float bia = b[c] - m[c] * scl;
    float v0 = any ? mx : 0.f;
    float y  = any ? fmaxf(v0 * scl + bia, 0.f) : 0.f;
    Out[((size_t)(c * OD + od) * H + h) * W + w] = y;
}

// ---------------------------------------------------------------------------
// Host launch
// ---------------------------------------------------------------------------
extern "C" void kernel_launch(void* const* d_in, const int* in_sizes, int n_in,
                              void* d_out, int out_size, void* d_ws, size_t ws_size,
                              hipStream_t stream) {
    const float* vf    = (const float*)d_in[0];
    const int*   coors = (const int*)d_in[1];
    const float* w_in  = (const float*)d_in[3];
    const float* w_up  = (const float*)d_in[4];
    const float* w_mid = (const float*)d_in[5];
    const float* bn0g  = (const float*)d_in[6];
    const float* bn0b  = (const float*)d_in[7];
    const float* bn0m  = (const float*)d_in[8];
    const float* bn0v  = (const float*)d_in[9];
    const float* bng   = (const float*)d_in[10];
    const float* bnb   = (const float*)d_in[11];
    const float* bnm   = (const float*)d_in[12];
    const float* bnv   = (const float*)d_in[13];
    float* out = (float*)d_out;

    const int N = in_sizes[0] / 16;

    // ---- carve workspace (256B-aligned chunks) ----
    char* ws = (char*)d_ws;
    size_t off = 0;
    auto alloc = [&](size_t bytes) -> void* {
        void* p = ws + off;
        off = (off + bytes + 255) & ~(size_t)255;
        return p;
    };
    const size_t P41 = (size_t)41 * 160 * 160;
    const size_t P21 = (size_t)21 * 160 * 160;
    const size_t P10 = (size_t)10 * 160 * 160;
    unsigned short* bufP = (unsigned short*)alloc(P41 * 64 * 2);
    unsigned short* bufQ = (unsigned short*)alloc(P41 * 64 * 2);
    unsigned char*  maskA = (unsigned char*)alloc(P41);
    unsigned char*  maskB = (unsigned char*)alloc(P21);
    unsigned char*  maskC = (unsigned char*)alloc(P10);
    unsigned short* WtIn  = (unsigned short*)alloc((size_t)32 * 27 * 32 * 2);
    unsigned short* WtUp  = (unsigned short*)alloc((size_t)64 * 27 * 32 * 2);
    unsigned short* WtMid = (unsigned short*)alloc((size_t)6 * 64 * 27 * 64 * 2);
    (void)ws_size; (void)n_in; (void)out_size;

    // ---- zero X0 (32ch region of bufP) and maskA; weights prep; scatter ----
    zero_u32<<<2048, 256, 0, stream>>>((unsigned*)bufP, (long long)(P41 * 32 * 2 / 4));
    zero_u32<<<256, 256, 0, stream>>>((unsigned*)maskA, (long long)(P41 / 4));
    scatter_kernel<<<(N + 255) / 256, 256, 0, stream>>>(vf, coors, bufP, maskA, N);
    prep_w<<<(32 * 27 * 32 + 255) / 256, 256, 0, stream>>>(w_in, WtIn, 16, 32, 32);
    prep_w<<<(64 * 27 * 32 + 255) / 256, 256, 0, stream>>>(w_up, WtUp, 32, 32, 64);
    for (int l = 0; l < 6; ++l)
        prep_w<<<(64 * 27 * 64 + 255) / 256, 256, 0, stream>>>(
            w_mid + (size_t)l * 27 * 64 * 64, WtMid + (size_t)l * 64 * 27 * 64, 64, 64, 64);

    const size_t WMSTRIDE = (size_t)64 * 27 * 64;

    // ---- conv_in 16(pad32)->32, D=41 ----
    conv3d_wmma<32, 32, 2><<<41 * 80, 128, 0, stream>>>(
        bufP, WtIn, maskA, bufQ, 41, bn0g, bn0b, bn0m, bn0v);
    // ---- conv_up 32->64, D=41, bn row 0 ----
    conv3d_wmma<32, 64, 1><<<41 * 160, 128, 0, stream>>>(
        bufQ, WtUp, maskA, bufP, 41, bng + 0, bnb + 0, bnm + 0, bnv + 0);
    // ---- pool1 (3,3,3)/(2,1,1)/pad(1,1,1): D 41->21, bn row 1 ----
    pool_bn<1><<<(int)((P21 * 64 + 255) / 256), 256, 0, stream>>>(
        bufP, maskA, bufQ, maskB, 41, 21, bng + 64, bnb + 64, bnm + 64, bnv + 64);
    // ---- mid convs 0..2 @ D=21, bn rows 2..4 ----
    conv3d_wmma<64, 64, 1><<<21 * 160, 128, 0, stream>>>(
        bufQ, WtMid + 0 * WMSTRIDE, maskB, bufP, 21, bng + 128, bnb + 128, bnm + 128, bnv + 128);
    conv3d_wmma<64, 64, 1><<<21 * 160, 128, 0, stream>>>(
        bufP, WtMid + 1 * WMSTRIDE, maskB, bufQ, 21, bng + 192, bnb + 192, bnm + 192, bnv + 192);
    conv3d_wmma<64, 64, 1><<<21 * 160, 128, 0, stream>>>(
        bufQ, WtMid + 2 * WMSTRIDE, maskB, bufP, 21, bng + 256, bnb + 256, bnm + 256, bnv + 256);
    // ---- pool2 (3,3,3)/(2,1,1)/pad(0,1,1): D 21->10, bn row 5 ----
    pool_bn<0><<<(int)((P10 * 64 + 255) / 256), 256, 0, stream>>>(
        bufP, maskB, bufQ, maskC, 21, 10, bng + 320, bnb + 320, bnm + 320, bnv + 320);
    // ---- mid convs 3..5 @ D=10, bn rows 6..8 ----
    conv3d_wmma<64, 64, 1><<<10 * 160, 128, 0, stream>>>(
        bufQ, WtMid + 3 * WMSTRIDE, maskC, bufP, 10, bng + 384, bnb + 384, bnm + 384, bnv + 384);
    conv3d_wmma<64, 64, 1><<<10 * 160, 128, 0, stream>>>(
        bufP, WtMid + 4 * WMSTRIDE, maskC, bufQ, 10, bng + 448, bnb + 448, bnm + 448, bnv + 448);
    conv3d_wmma<64, 64, 1><<<10 * 160, 128, 0, stream>>>(
        bufQ, WtMid + 5 * WMSTRIDE, maskC, bufP, 10, bng + 512, bnb + 512, bnm + 512, bnv + 512);
    // ---- pool3 (3,1,1)/(2,1,1): D 10->4, bn row 9, transpose to (c*4+d,h,w) f32 ----
    pool3_out<<<(int)(((size_t)4 * 160 * 160 * 64 + 255) / 256), 256, 0, stream>>>(
        bufP, maskC, out, bng + 576, bnb + 576, bnm + 576, bnv + 576);
}
